// EnhancedMultiHeadAttention_89283780149856
// MI455X (gfx1250) — matrix operand order repaired and checked by reading, not compiled, not run on hardware
//
#include <hip/hip_runtime.h>

typedef __attribute__((ext_vector_type(16))) _Float16 v16h;
typedef __attribute__((ext_vector_type(8)))  _Float16 v8h;
typedef __attribute__((ext_vector_type(8)))  float    v8f;
typedef __attribute__((ext_vector_type(4)))  int      v4i;

#define WMMA_F32_F16(a, b, c) \
  __builtin_amdgcn_wmma_f32_16x16x32_f16(false, (a), false, (b), (short)0, (c), false, false)

// ---------------------------------------------------------------------------
// Async global->LDS copy (CDNA5 GLOBAL_LOAD_ASYNC_TO_LDS_B128, ASYNCcnt),
// guarded so the build falls back to a synchronous copy if the toolchain
// lacks the gfx1250 builtins.  (Round-2 disasm confirmed the async path.)
// ---------------------------------------------------------------------------
#if defined(__has_builtin)
#  if __has_builtin(__builtin_amdgcn_global_load_async_to_lds_b128) && \
      __has_builtin(__builtin_amdgcn_s_wait_asynccnt)
#    define HAS_ASYNC_LDS 1
#  endif
#endif
#ifndef HAS_ASYNC_LDS
#  define HAS_ASYNC_LDS 0
#endif

#define GAS __attribute__((address_space(1)))
#define LAS __attribute__((address_space(3)))

__device__ __forceinline__ void async_copy_b128(const _Float16* g, _Float16* lds) {
#if HAS_ASYNC_LDS
  GAS v4i* gp = (GAS v4i*)(unsigned long long)g;                 // generic global == AS1 numerically
  LAS v4i* lp = (LAS v4i*)(unsigned int)(unsigned long long)lds; // low 32 bits of generic LDS ptr = LDS offset
  __builtin_amdgcn_global_load_async_to_lds_b128(gp, lp, 0, 0);
#else
  *(v8h*)lds = *(const v8h*)g;
#endif
}

__device__ __forceinline__ void wait_async0() {
#if HAS_ASYNC_LDS
  __builtin_amdgcn_s_wait_asynccnt(0);
#endif
}

// ---------------------------------------------------------------------------
// WMMA fragment loaders (wave32, 16x16x32 f16 -> f32)
// A 16x32 (MxK): lane L holds row M=L&15; K runs of 8 at kb=(L>>4)*8 and kb+16.
// B as B^T[N][K] row-major: lane L holds col N=L&15; K runs of 8 at
// kb=(L>>4)*8 and kb+16. C/D: VGPR r <-> row r+8*(L>>4), col = L&15.
// ---------------------------------------------------------------------------
__device__ __forceinline__ v16h load_a_frag(const _Float16* p, int ld, int m0, int k0) {
  const int lane = threadIdx.x & 31;
  const _Float16* r = p + (m0 + (lane & 15)) * ld + k0 + ((lane >> 4) << 3);
  v8h lo = *(const v8h*)(r);
  v8h hv = *(const v8h*)(r + 16);
  v16h out;
#pragma unroll
  for (int i = 0; i < 8; ++i) { out[i] = lo[i]; out[i + 8] = hv[i]; }
  return out;
}

__device__ __forceinline__ v16h load_bt_frag(const _Float16* p, int ld, int n0, int k0) {
  const int lane = threadIdx.x & 31;
  const _Float16* r = p + (n0 + (lane & 15)) * ld + k0 + ((lane >> 4) << 3);
  v8h lo = *(const v8h*)(r);
  v8h hv = *(const v8h*)(r + 16);
  v16h out;
#pragma unroll
  for (int i = 0; i < 8; ++i) { out[i] = lo[i]; out[i + 8] = hv[i]; }
  return out;
}

// ---------------------------------------------------------------------------
// f32 -> f16 cast, 8 elements per thread
// ---------------------------------------------------------------------------
__global__ void cast_f32_to_f16(const float* __restrict__ src,
                                _Float16* __restrict__ dst, long n8) {
  long i = (long)blockIdx.x * blockDim.x + threadIdx.x;
  if (i >= n8) return;
  const float4* s = (const float4*)src + i * 2;
  float4 a = s[0], b = s[1];
  v8h o;
  o[0] = (_Float16)a.x; o[1] = (_Float16)a.y; o[2] = (_Float16)a.z; o[3] = (_Float16)a.w;
  o[4] = (_Float16)b.x; o[5] = (_Float16)b.y; o[6] = (_Float16)b.z; o[7] = (_Float16)b.w;
  *((v8h*)dst + i) = o;
}

// ---------------------------------------------------------------------------
// V transpose: vh[B*S, 64] -> vt[B, 64, S]  (one-time, LDS-tiled)
// Grid: (S/64, B), 256 threads.
// ---------------------------------------------------------------------------
__global__ void __launch_bounds__(256)
transpose_v(const _Float16* __restrict__ vh, _Float16* __restrict__ vt, int S) {
  __shared__ _Float16 tile[64][72];
  const int b = blockIdx.y;
  const int t0 = blockIdx.x * 64;
  const int tid = threadIdx.x;
  const int r = tid >> 2, c0 = (tid & 3) * 16;
  const _Float16* g = vh + (long)(b * S + t0 + r) * 64 + c0;
  v8h x0 = *(const v8h*)g;
  v8h x1 = *(const v8h*)(g + 8);
#pragma unroll
  for (int e = 0; e < 8; ++e) { tile[r][c0 + e] = x0[e]; tile[r][c0 + 8 + e] = x1[e]; }
  __syncthreads();
  const int d = tid >> 2, tc0 = (tid & 3) * 16;
  v8h o0, o1;
#pragma unroll
  for (int e = 0; e < 8; ++e) { o0[e] = tile[tc0 + e][d]; o1[e] = tile[tc0 + 8 + e][d]; }
  _Float16* o = vt + ((long)b * 64 + d) * S + t0 + tc0;
  *(v8h*)o = o0;
  *(v8h*)(o + 8) = o1;
}

// ---------------------------------------------------------------------------
// C[M,N] = A[M,K] @ W[N,K]^T   (f16 in, OutT out)
// Block: 128 threads = 4 waves; tile 128x64; K stepped by 32, double-buffered
// async LDS staging. M multiple of 128, N multiple of 64, K multiple of 32.
// ---------------------------------------------------------------------------
template <typename OutT>
__global__ void __launch_bounds__(128)
gemm_xwt(const _Float16* __restrict__ A, const _Float16* __restrict__ W,
         OutT* __restrict__ C, int M, int N, int K) {
  constexpr int ALD = 40;  // 80B stride -> conflict-free b128 LDS loads
  __shared__ _Float16 As[2][128 * ALD];
  __shared__ _Float16 Bs[2][64 * ALD];

  const int m0 = blockIdx.y * 128;
  const int n0 = blockIdx.x * 64;
  const int tid = threadIdx.x;
  const int lane = tid & 31, wv = tid >> 5;
  const int hi = lane >> 4, col = lane & 15;

  auto stage = [&](int buf, int k0) {
    const _Float16* ga = A + (long)(m0 + tid) * K + k0;        // 32 halves / thread
    _Float16* la = &As[buf][tid * ALD];
#pragma unroll
    for (int j = 0; j < 4; ++j) async_copy_b128(ga + j * 8, la + j * 8);
    const _Float16* gw = W + (long)(n0 + (tid >> 1)) * K + k0 + (tid & 1) * 16;
    _Float16* lb = &Bs[buf][(tid >> 1) * ALD + (tid & 1) * 16];
#pragma unroll
    for (int j = 0; j < 2; ++j) async_copy_b128(gw + j * 8, lb + j * 8);
  };

  stage(0, 0);
  wait_async0();
  __syncthreads();

  v8f acc[2][4] = {};
  int buf = 0;
  for (int k0 = 0; k0 < K; k0 += 32) {
    if (k0 + 32 < K) stage(buf ^ 1, k0 + 32);  // overlap copy with WMMA

    v16h a0 = load_a_frag(&As[buf][0], ALD, wv * 32, 0);
    v16h a1 = load_a_frag(&As[buf][0], ALD, wv * 32 + 16, 0);
#pragma unroll
    for (int n = 0; n < 4; ++n) {
      v16h b = load_bt_frag(&Bs[buf][0], ALD, n * 16, 0);
      acc[0][n] = WMMA_F32_F16(a0, b, acc[0][n]);
      acc[1][n] = WMMA_F32_F16(a1, b, acc[1][n]);
    }
    wait_async0();
    __syncthreads();
    buf ^= 1;
  }

#pragma unroll
  for (int mi = 0; mi < 2; ++mi)
#pragma unroll
    for (int n = 0; n < 4; ++n)
#pragma unroll
      for (int r = 0; r < 8; ++r) {
        long row = m0 + wv * 32 + mi * 16 + r + 8 * hi;
        C[row * N + n0 + n * 16 + col] = (OutT)acc[mi][n][r];
      }
}

// ---------------------------------------------------------------------------
// Flash attention, multi-query (shared K/V), head dim 64.
// Block: 128 threads = 4 waves; 64 query rows per block, one (head, batch).
// K and V^T tiles double-buffered via async global->LDS; softmax in base-2.
// VT is the pre-transposed V: vt[b][d][t].
// ---------------------------------------------------------------------------
__global__ void __launch_bounds__(128)
flash_attn(const _Float16* __restrict__ Qh, const _Float16* __restrict__ Kh,
           const _Float16* __restrict__ VT, _Float16* __restrict__ Oh,
           const float* __restrict__ scale_p, int S, int D) {
  constexpr int LD = 72;  // 144B stride -> conflict-free b128 LDS loads
  __shared__ _Float16 Qs[64 * LD];
  __shared__ _Float16 Ks[2][64 * LD];
  __shared__ _Float16 VTs[2][64 * LD];
  __shared__ _Float16 Ps[4][16 * LD];

  const int s0 = blockIdx.x * 64;
  const int h  = blockIdx.y;
  const int b  = blockIdx.z;
  const int tid = threadIdx.x;
  const int lane = tid & 31, wv = tid >> 5;
  const int hi = lane >> 4, col = lane & 15;
  const float c2 = scale_p[0] * 1.44269504088896f;  // scale * log2(e)

  const int r = tid >> 1, c0 = (tid & 1) * 32;

  auto stage_kv = [&](int buf, int t0) {
    const _Float16* gk = Kh + (long)(b * S + t0 + r) * 64 + c0;
    _Float16* lk = &Ks[buf][r * LD + c0];
#pragma unroll
    for (int j = 0; j < 4; ++j) async_copy_b128(gk + j * 8, lk + j * 8);
    const _Float16* gv = VT + ((long)b * 64 + r) * S + t0 + c0;
    _Float16* lv = &VTs[buf][r * LD + c0];
#pragma unroll
    for (int j = 0; j < 4; ++j) async_copy_b128(gv + j * 8, lv + j * 8);
  };

  // Stage Q tile (async, once) + first K/V tile.
  {
    const _Float16* g = Qh + (long)(b * S + s0 + r) * D + h * 64 + c0;
#pragma unroll
    for (int j = 0; j < 4; ++j) async_copy_b128(g + j * 8, Qs + r * LD + c0 + j * 8);
  }
  stage_kv(0, 0);
  wait_async0();
  __syncthreads();

  const v16h aq0 = load_a_frag(Qs, LD, wv * 16, 0);
  const v16h aq1 = load_a_frag(Qs, LD, wv * 16, 32);

  float mrow[8], lrow[8];
  v8f oacc[4] = {};
#pragma unroll
  for (int rr = 0; rr < 8; ++rr) { mrow[rr] = -1e30f; lrow[rr] = 0.f; }

  int buf = 0;
  for (int t0 = 0; t0 < S; t0 += 64) {
    if (t0 + 64 < S) stage_kv(buf ^ 1, t0 + 64);  // overlap copy with compute

    // scores S = Q @ K^T (K-dim 64 = two WMMA k-steps), pre-scaled by log2e*scale
    v8f sc[4];
#pragma unroll
    for (int n = 0; n < 4; ++n) {
      v8f c = {};
      c = WMMA_F32_F16(aq0, load_bt_frag(&Ks[buf][0], LD, n * 16, 0), c);
      c = WMMA_F32_F16(aq1, load_bt_frag(&Ks[buf][0], LD, n * 16, 32), c);
#pragma unroll
      for (int rr = 0; rr < 8; ++rr) c[rr] *= c2;
      sc[n] = c;
    }

    // Online softmax (base 2); each lane owns 8 rows (r + 8*hi); reduce across
    // the 16-lane half with a 4-stage xor butterfly (stays within the half).
    float newm[8];
#pragma unroll
    for (int rr = 0; rr < 8; ++rr) {
      float m = mrow[rr];
#pragma unroll
      for (int n = 0; n < 4; ++n) m = fmaxf(m, sc[n][rr]);
      newm[rr] = m;
    }
#pragma unroll
    for (int mask = 1; mask <= 8; mask <<= 1)
#pragma unroll
      for (int rr = 0; rr < 8; ++rr)
        newm[rr] = fmaxf(newm[rr], __shfl_xor(newm[rr], mask, 32));

    float rsum[8];
#pragma unroll
    for (int rr = 0; rr < 8; ++rr) rsum[rr] = 0.f;
#pragma unroll
    for (int n = 0; n < 4; ++n)
#pragma unroll
      for (int rr = 0; rr < 8; ++rr) {
        float p = exp2f(sc[n][rr] - newm[rr]);
        sc[n][rr] = p;
        rsum[rr] += p;
      }
#pragma unroll
    for (int mask = 1; mask <= 8; mask <<= 1)
#pragma unroll
      for (int rr = 0; rr < 8; ++rr)
        rsum[rr] += __shfl_xor(rsum[rr], mask, 32);

#pragma unroll
    for (int rr = 0; rr < 8; ++rr) {
      float corr = exp2f(mrow[rr] - newm[rr]);
      mrow[rr] = newm[rr];
      lrow[rr] = lrow[rr] * corr + rsum[rr];
#pragma unroll
      for (int n = 0; n < 4; ++n) oacc[n][rr] *= corr;
    }

    // Re-layout P (C/D fragment) -> A fragment via per-wave LDS patch.
    _Float16* Pw = &Ps[wv][0];
#pragma unroll
    for (int n = 0; n < 4; ++n)
#pragma unroll
      for (int rr = 0; rr < 8; ++rr)
        Pw[(rr + 8 * hi) * LD + n * 16 + col] = (_Float16)sc[n][rr];

    v16h p0 = load_a_frag(Pw, LD, 0, 0);
    v16h p1 = load_a_frag(Pw, LD, 0, 32);
#pragma unroll
    for (int n = 0; n < 4; ++n) {
      oacc[n] = WMMA_F32_F16(p0, load_bt_frag(&VTs[buf][0], LD, n * 16, 0), oacc[n]);
      oacc[n] = WMMA_F32_F16(p1, load_bt_frag(&VTs[buf][0], LD, n * 16, 32), oacc[n]);
    }

    wait_async0();
    __syncthreads();
    buf ^= 1;
  }

  // Normalize and write attention output (f16) for the final projection.
#pragma unroll
  for (int n = 0; n < 4; ++n)
#pragma unroll
    for (int rr = 0; rr < 8; ++rr) {
      long row = (long)b * S + s0 + wv * 16 + rr + 8 * hi;
      Oh[row * D + h * 64 + n * 16 + col] = (_Float16)(oacc[n][rr] / lrow[rr]);
    }
}

// ---------------------------------------------------------------------------
// Launcher
// ---------------------------------------------------------------------------
extern "C" void kernel_launch(void* const* d_in, const int* in_sizes, int n_in,
                              void* d_out, int out_size, void* d_ws, size_t ws_size,
                              hipStream_t stream) {
  (void)in_sizes; (void)n_in; (void)out_size; (void)ws_size;
  const int B = 2, S = 2048, D = 1024, H = 16, dk = 64;
  const long MS = (long)B * S;  // 4096 total rows

  const float* x     = (const float*)d_in[0];
  const float* w_q   = (const float*)d_in[1];
  const float* w_k   = (const float*)d_in[2];
  const float* w_v   = (const float*)d_in[3];
  const float* w_out = (const float*)d_in[4];
  const float* scale = (const float*)d_in[5];
  float* out = (float*)d_out;

  char* ws = (char*)d_ws;
  size_t off = 0;
  auto alloc = [&](size_t bytes) -> void* {
    void* p = ws + off;
    off = (off + bytes + 255) & ~(size_t)255;
    return p;
  };
  _Float16* xh  = (_Float16*)alloc((size_t)MS * D * 2);   // x in f16
  _Float16* qh  = (_Float16*)alloc((size_t)MS * D * 2);   // Q = x @ Wq^T
  _Float16* ah  = (_Float16*)alloc((size_t)MS * D * 2);   // attention output
  _Float16* kh  = (_Float16*)alloc((size_t)MS * dk * 2);  // shared K
  _Float16* vh  = (_Float16*)alloc((size_t)MS * dk * 2);  // shared V
  _Float16* vt  = (_Float16*)alloc((size_t)B * dk * S * 2); // V transposed [b][d][t]
  _Float16* wqh = (_Float16*)alloc((size_t)D * D * 2);
  _Float16* wkh = (_Float16*)alloc((size_t)dk * D * 2);
  _Float16* wvh = (_Float16*)alloc((size_t)dk * D * 2);
  _Float16* woh = (_Float16*)alloc((size_t)D * D * 2);

  auto cast = [&](const float* s, _Float16* d, long n) {
    long n8 = n / 8;
    int blocks = (int)((n8 + 255) / 256);
    cast_f32_to_f16<<<blocks, 256, 0, stream>>>(s, d, n8);
  };
  cast(x, xh, MS * D);
  cast(w_q, wqh, (long)D * D);
  cast(w_k, wkh, (long)dk * D);
  cast(w_v, wvh, (long)dk * D);
  cast(w_out, woh, (long)D * D);

  // Projections
  gemm_xwt<_Float16><<<dim3(D / 64, MS / 128), 128, 0, stream>>>(xh, wqh, qh, (int)MS, D, D);
  gemm_xwt<_Float16><<<dim3(1,      MS / 128), 128, 0, stream>>>(xh, wkh, kh, (int)MS, dk, D);
  gemm_xwt<_Float16><<<dim3(1,      MS / 128), 128, 0, stream>>>(xh, wvh, vh, (int)MS, dk, D);

  // One-time V transpose: vh[B*S,64] -> vt[B,64,S]
  transpose_v<<<dim3(S / 64, B), 256, 0, stream>>>(vh, vt, S);

  // Attention
  flash_attn<<<dim3(S / 64, H, B), 128, 0, stream>>>(qh, kh, vt, ah, scale, S, D);

  // Output projection (f32 result)
  gemm_xwt<float><<<dim3(D / 64, MS / 128), 128, 0, stream>>>(ah, woh, out, (int)MS, D, D);
}
